// Block_65395172049024
// MI455X (gfx1250) — matrix operand order repaired and checked by reading, not compiled, not run on hardware
//
#include <hip/hip_runtime.h>
#include <hip/hip_bf16.h>

// ---------------------------------------------------------------------------
// Fused windowed-attention block for MI455X (gfx1250, wave32, WMMA).
// One workgroup (256 thr = 8 waves) per window m=(n,t); entire block fused:
// gather -> LN1 -> q/k/v/t/o GEMMs + head softmax pooling -> window-agg GEMM
// -> LN2 -> FFN GEMM + GELU + FFN GEMM -> residual -> global store.
// All GEMMs via v_wmma_f32_16x16x32_f16. Weights converted/transposed to f16
// in d_ws once so B fragments are contiguous b128 loads.
// ---------------------------------------------------------------------------

typedef __attribute__((ext_vector_type(16))) _Float16 v16h;
typedef __attribute__((ext_vector_type(8)))  _Float16 v8h;
typedef __attribute__((ext_vector_type(8)))  float    v8f;

constexpr int LDAH = 136;   // f16 tile row stride (halves) -> 272B, kills bank conflicts
constexpr int LDXS = 132;   // f32 staging row stride (floats)

// f16 weight arena offsets (in halves) inside d_ws
constexpr int WQ_T  = 0;
constexpr int WK_T  = 16384;
constexpr int WV_T  = 32768;
constexpr int WT_T  = 49152;
constexpr int WO_T  = 65536;
constexpr int F1_T  = 81920;
constexpr int F2_T  = 98304;
constexpr int WQA_T = 114688;  // [16][128], cols>=8 zero
constexpr int WKA_T = 116736;
constexpr int AGG_T = 118784;  // [128][640]
constexpr int W16_TOTAL = 200704;

// dynamic-LDS layout (bytes, all 16B aligned)
constexpr unsigned OFF_XS   = 0;                    // f32 [128][132] = 67584 (stage; reused below)
constexpr unsigned OFF_ATT  = 0;                    // f16 [128][136] = 34816 (after attention)
constexpr unsigned OFF_AOUT = 34816;                // f32 [32][132]  = 16896 (att_out, residual)
constexpr unsigned OFF_NX   = 67584;                // f16 tile (nx, later kvt)
constexpr unsigned OFF_Q    = 102400;               // f16 tile
constexpr unsigned OFF_K    = 137216;               // f16 tile
constexpr unsigned OFF_V    = 172032;               // f16 tile (becomes kv in place)
constexpr unsigned OFF_SC   = 206848;               // f32 [128][16] head scores
constexpr unsigned OFF_HW   = 215040;               // f32 [8][128] softmax weights
constexpr unsigned OFF_PQ   = 219136;               // f32 [128] pooled_q (c = h*16+hc)
constexpr unsigned OFF_PK   = 219648;               // f32 [128] pooled_k
constexpr unsigned OFF_LN2  = 220160;               // f16 [32][136]
constexpr unsigned OFF_H1   = 228864;               // f16 [32][136]
constexpr unsigned SMEM_TOTAL = 237568;             // ~232 KB of the 320 KB WGP LDS

__device__ __forceinline__ v16h pack16(v8h lo, v8h hi) {
  v16h r;
#pragma unroll
  for (int i = 0; i < 8; ++i) { r[i] = lo[i]; r[i + 8] = hi[i]; }
  return r;
}

// A fragment, 16x32 f16, row-major LDS tile. Per ISA: lane holds row=row0+(lane&15),
// halves 0..7 = K kb..kb+7, halves 8..15 = K kb+16..kb+23, kb = k0 + (lane>=16)*8.
__device__ __forceinline__ v16h lda16x32(const _Float16* base, int ld, int row, int k0, int lane) {
  const _Float16* p = base + row * ld + k0 + ((lane >> 4) << 3);
  return pack16(*(const v8h*)p, *(const v8h*)(p + 16));
}

// Same, but each element scaled by per-channel gate g[k] (kq = k * pooled_q).
__device__ __forceinline__ v16h lda16x32_gated(const _Float16* base, int ld, int row, int k0,
                                               int lane, const float* g) {
  int kb = k0 + ((lane >> 4) << 3);
  const _Float16* p = base + row * ld + kb;
  v16h r;
#pragma unroll
  for (int i = 0; i < 8; ++i) {
    r[i]     = (_Float16)((float)p[i]      * g[kb + i]);
    r[i + 8] = (_Float16)((float)p[16 + i] * g[kb + 16 + i]);
  }
  return r;
}

// B fragment, 32x16 f16, from weights stored TRANSPOSED [N_out][K]: lane holds
// col = col0+(lane&15), halves 0..7 = K kb..kb+7, halves 8..15 = K kb+16..kb+23, kb = k0 + (lane>=16)*8.
__device__ __forceinline__ v16h ldb32x16(const _Float16* Bt, int ld, int col0, int k0, int lane) {
  const _Float16* p = Bt + (size_t)(col0 + (lane & 15)) * ld + k0 + ((lane >> 4) << 3);
  return pack16(*(const v8h*)p, *(const v8h*)(p + 16));
}

__device__ __forceinline__ v8f wmma_f16(v16h a, v16h b, v8f c) {
  return __builtin_amdgcn_wmma_f32_16x16x32_f16(false, a, false, b, (short)0, c, false, false);
}

// 128x128 = [128x128]x[128x128] GEMM; wave owns row-tile rt=wave; f16 out to LDS.
__device__ void gemm128(const _Float16* A, const _Float16* Bt, const float* bias,
                        _Float16* outT, const _Float16* addT, int wave, int lane) {
  const int rt = wave, cl = lane & 15, rs = (lane >> 4) << 3;
  for (int ct = 0; ct < 8; ++ct) {
    v8f acc = {};
#pragma unroll
    for (int kk = 0; kk < 4; ++kk) {
      v16h a = lda16x32(A, LDAH, rt * 16 + cl, kk * 32, lane);
      v16h b = ldb32x16(Bt, 128, ct * 16, kk * 32, lane);
      acc = wmma_f16(a, b, acc);
    }
    const int col = ct * 16 + cl;
    const float bb = bias ? bias[col] : 0.f;
    const int r0 = rt * 16 + rs;
#pragma unroll
    for (int i = 0; i < 8; ++i) {
      float v = acc[i] + bb;
      if (addT) v += (float)addT[(r0 + i) * LDAH + col];
      outT[(r0 + i) * LDAH + col] = (_Float16)v;
    }
  }
}

// [128x128]x[128x16] head-score GEMM (wqa/wka padded to 16 cols); fp32 out, *SCALE.
__device__ void gemm_heads(const _Float16* A, const _Float16* Wt, const float* bias8,
                           float* sc, const float* gate, int wave, int lane) {
  const int rt = wave, cl = lane & 15, rs = (lane >> 4) << 3;
  v8f acc = {};
#pragma unroll
  for (int kk = 0; kk < 4; ++kk) {
    v16h a = gate ? lda16x32_gated(A, LDAH, rt * 16 + cl, kk * 32, lane, gate)
                  : lda16x32(A, LDAH, rt * 16 + cl, kk * 32, lane);
    v16h b = ldb32x16(Wt, 128, 0, kk * 32, lane);
    acc = wmma_f16(a, b, acc);
  }
  const float bb = (cl < 8) ? bias8[cl] : 0.f;
  const int r0 = rt * 16 + rs;
#pragma unroll
  for (int i = 0; i < 8; ++i) sc[(r0 + i) * 16 + cl] = (acc[i] + bb) * 0.25f;  // SCALE=1/sqrt(16)
}

// wave h: softmax over l<125 of sc[:,h], then pooled[h*16+hc] = sum_l w[l]*key[l][h*16+hc]
__device__ void softmax_pool(const float* sc, const _Float16* keyT, float* hw, float* pooled,
                             int wave, int lane) {
  const int h = wave;
  float mx = -3.0e38f;
  for (int l = lane; l < 125; l += 32) mx = fmaxf(mx, sc[l * 16 + h]);
#pragma unroll
  for (int o = 16; o; o >>= 1) mx = fmaxf(mx, __shfl_xor(mx, o, 32));
  float sum = 0.f;
  for (int l = lane; l < 125; l += 32) {
    float e = __expf(sc[l * 16 + h] - mx);
    hw[h * 128 + l] = e;
    sum += e;
  }
#pragma unroll
  for (int o = 16; o; o >>= 1) sum += __shfl_xor(sum, o, 32);
  const float inv = 1.f / sum;
  if (lane < 16) {
    float acc = 0.f;
    for (int l = 0; l < 125; ++l)
      acc += hw[h * 128 + l] * (float)keyT[l * LDAH + h * 16 + lane];
    pooled[h * 16 + lane] = acc * inv;
  }
}

// window aggregation: out[v][c_out] = sum_{w,c} att[(w*25+v)][c] * aggT[c_out][w*128+c]
// K = 640 (20 WMMA steps); + agg_b + x residual -> fp32 att_out.
__device__ void agg_fn(const _Float16* att, const _Float16* aggT, const float* aggB,
                       const float* x, float* aout, int n, int t, int wave, int lane) {
  const int ct = wave, cl = lane & 15, rs = (lane >> 4) << 3;
  const int col = ct * 16 + cl;
  for (int rt = 0; rt < 2; ++rt) {
    const int vl = rt * 16 + cl;
    const int vrow = (vl < 25) ? vl : 0;  // clamp pad rows (outputs discarded)
    v8f acc = {};
    for (int kk = 0; kk < 20; ++kk) {
      const int w = kk >> 2, c0 = (kk & 3) << 5;
      v16h a = lda16x32(att, LDAH, w * 25 + vrow, c0, lane);
      v16h b = ldb32x16(aggT, 640, ct * 16, kk * 32, lane);
      acc = wmma_f16(a, b, acc);
    }
    const float bb = aggB[col];
    const float* xp = x + (((size_t)n * 128 + col) * 120 + t) * 25;
    const int v0 = rt * 16 + rs;
#pragma unroll
    for (int i = 0; i < 8; ++i) {
      const int v = v0 + i;
      if (v < 25) aout[v * LDXS + col] = acc[i] + bb + xp[v];
    }
  }
}

// FFN GEMM1 [32x128]x[128x128] with tanh-GELU epilogue -> f16.
__device__ void ffn1_fn(const _Float16* A, const _Float16* Bt, const float* bias,
                        _Float16* o16, int wave, int lane) {
  const int ct = wave, cl = lane & 15, rs = (lane >> 4) << 3;
  for (int rt = 0; rt < 2; ++rt) {
    v8f acc = {};
#pragma unroll
    for (int kk = 0; kk < 4; ++kk) {
      v16h a = lda16x32(A, LDAH, rt * 16 + cl, kk * 32, lane);
      v16h b = ldb32x16(Bt, 128, ct * 16, kk * 32, lane);
      acc = wmma_f16(a, b, acc);
    }
    const int col = ct * 16 + cl;
    const float bb = bias[col];
    const int r0 = rt * 16 + rs;
#pragma unroll
    for (int i = 0; i < 8; ++i) {
      float xg = acc[i] + bb;
      float g = 0.5f * xg * (1.f + tanhf(0.7978845608f * (xg + 0.044715f * xg * xg * xg)));
      o16[(r0 + i) * LDAH + col] = (_Float16)g;
    }
  }
}

// FFN GEMM2 + bias + att_out residual -> global out [N,C,T,V].
__device__ void ffn2_fn(const _Float16* A, const _Float16* Bt, const float* bias,
                        const float* aout, float* out, int n, int t, int wave, int lane) {
  const int ct = wave, cl = lane & 15, rs = (lane >> 4) << 3;
  for (int rt = 0; rt < 2; ++rt) {
    v8f acc = {};
#pragma unroll
    for (int kk = 0; kk < 4; ++kk) {
      v16h a = lda16x32(A, LDAH, rt * 16 + cl, kk * 32, lane);
      v16h b = ldb32x16(Bt, 128, ct * 16, kk * 32, lane);
      acc = wmma_f16(a, b, acc);
    }
    const int col = ct * 16 + cl;
    const float bb = bias[col];
    const int v0 = rt * 16 + rs;
    float* op = out + (((size_t)n * 128 + col) * 120 + t) * 25;
#pragma unroll
    for (int i = 0; i < 8; ++i) {
      const int v = v0 + i;
      if (v < 25) op[v] = acc[i] + bb + aout[v * LDXS + col];
    }
  }
}

// ---------------------------------------------------------------------------
// one-shot weight repack: fp32 -> f16, transposed to [C_out][K] for B fragments
// ---------------------------------------------------------------------------
__global__ void convert_weights(const float* __restrict__ wq, const float* __restrict__ wk,
                                const float* __restrict__ wv, const float* __restrict__ wt,
                                const float* __restrict__ wo, const float* __restrict__ f1,
                                const float* __restrict__ f2, const float* __restrict__ wqa,
                                const float* __restrict__ wka, const float* __restrict__ aggw,
                                _Float16* __restrict__ out) {
  const int idx = blockIdx.x * blockDim.x + threadIdx.x;
  if (idx < 7 * 16384) {
    const int which = idx >> 14, r = idx & 16383;
    const int o = r >> 7, i = r & 127;
    const float* W;
    switch (which) {
      case 0: W = wq; break; case 1: W = wk; break; case 2: W = wv; break;
      case 3: W = wt; break; case 4: W = wo; break; case 5: W = f1; break;
      default: W = f2; break;
    }
    out[which * 16384 + o * 128 + i] = (_Float16)W[i * 128 + o];
  } else if (idx < 7 * 16384 + 4096) {
    const int r = idx - 7 * 16384;
    const int which = r >> 11, rr = r & 2047;
    const int o = rr >> 7, i = rr & 127;
    const float* W = which ? wka : wqa;
    const float v = (o < 8) ? W[i * 8 + o] : 0.f;
    out[WQA_T + which * 2048 + o * 128 + i] = (_Float16)v;
  } else if (idx < W16_TOTAL) {
    const int r = idx - AGG_T;
    const int o = r / 640, k = r - o * 640;
    const int w = k >> 7, i = k & 127;
    out[AGG_T + o * 640 + k] = (_Float16)aggw[(o * 128 + i) * 5 + w];  // agg_w[o][i][w]
  }
}

// ---------------------------------------------------------------------------
// fused block kernel: one workgroup per window m = (n, t)
// ---------------------------------------------------------------------------
__global__ __launch_bounds__(256, 1)
void fused_block_kernel(const float* __restrict__ x,
                        const float* __restrict__ g1, const float* __restrict__ b1,
                        const float* __restrict__ bq, const float* __restrict__ bqa,
                        const float* __restrict__ bk, const float* __restrict__ bka,
                        const float* __restrict__ bv, const float* __restrict__ bt,
                        const float* __restrict__ bo, const float* __restrict__ aggb,
                        const float* __restrict__ g2, const float* __restrict__ b2,
                        const float* __restrict__ fb1, const float* __restrict__ fb2,
                        const _Float16* __restrict__ w16, float* __restrict__ out) {
  extern __shared__ __align__(16) char smem[];
  const int tid = threadIdx.x, lane = tid & 31, wave = tid >> 5;
  const int m = blockIdx.x, n = m / 120, t = m - n * 120;

  float*    xs   = (float*)(smem + OFF_XS);
  _Float16* att  = (_Float16*)(smem + OFF_ATT);
  float*    aout = (float*)(smem + OFF_AOUT);
  _Float16* nx   = (_Float16*)(smem + OFF_NX);   // nx, later kvt
  _Float16* qt   = (_Float16*)(smem + OFF_Q);
  _Float16* kt   = (_Float16*)(smem + OFF_K);
  _Float16* vt   = (_Float16*)(smem + OFF_V);    // v, then kv in place
  float*    sc   = (float*)(smem + OFF_SC);
  float*    hw   = (float*)(smem + OFF_HW);
  float*    pq   = (float*)(smem + OFF_PQ);
  float*    pk   = (float*)(smem + OFF_PK);
  _Float16* ln2h = (_Float16*)(smem + OFF_LN2);
  _Float16* h1   = (_Float16*)(smem + OFF_H1);

  // ---- gather window tile: xs[(w*25+v)][c] = x[n,c,t+w-2,v] (zero-padded) ----
  for (int p = tid; p < 640; p += 256) {
    const int c = p / 5, w = p - c * 5;
    const int tw = t + w - 2;
    const bool ok = (unsigned)tw < 120u;
    const float* src = x + (((size_t)n * 128 + c) * 120 + (ok ? tw : 0)) * 25;
    float* dst = xs + (w * 25) * LDXS + c;
#pragma unroll 5
    for (int v = 0; v < 25; ++v) dst[v * LDXS] = ok ? src[v] : 0.f;
  }
  __syncthreads();

  // ---- LN1 over C per token; 2 lanes per row; f16 nx tile (pad rows -> 0) ----
  {
    const int row = wave * 16 + (lane >> 1);
    const int half = lane & 1;
    const float* rp = xs + row * LDXS + half * 64;
    float s = 0.f, s2 = 0.f;
    if (row < 125) {
#pragma unroll
      for (int j = 0; j < 64; ++j) { const float v = rp[j]; s += v; s2 += v * v; }
    }
    s += __shfl_xor(s, 1, 32);
    s2 += __shfl_xor(s2, 1, 32);
    const float mean = s * (1.f / 128.f);
    const float var = s2 * (1.f / 128.f) - mean * mean;
    const float inv = rsqrtf(var + 1e-5f);
    _Float16* op = nx + row * LDAH + half * 64;
    const float* gg = g1 + half * 64;
    const float* bb = b1 + half * 64;
#pragma unroll
    for (int j = 0; j < 64; ++j) {
      const float v = (row < 125) ? (rp[j] - mean) * inv * gg[j] + bb[j] : 0.f;
      op[j] = (_Float16)v;
    }
  }
  __syncthreads();

  // ---- q = nx@wq+bq ; head scores qa = (q'... actually nx@wqa+bqa)*SCALE ----
  gemm128(nx, w16 + WQ_T, bq, qt, nullptr, wave, lane);
  gemm_heads(nx, w16 + WQA_T, bqa, sc, nullptr, wave, lane);
  __syncthreads();
  softmax_pool(sc, qt, hw, pq, wave, lane);   // pooled_q
  __syncthreads();

  // ---- k = nx@wk+bk ; kq = k*pooled_q (gated frag) -> scores ----
  gemm128(nx, w16 + WK_T, bk, kt, nullptr, wave, lane);
  __syncthreads();
  gemm_heads(kt, w16 + WKA_T, bka, sc, pq, wave, lane);
  __syncthreads();
  softmax_pool(sc, kt, hw, pk, wave, lane);   // pooled_k over ungated k
  __syncthreads();

  // ---- v = nx@wv+bv ; kv = v*pooled_k in place ----
  gemm128(nx, w16 + WV_T, bv, vt, nullptr, wave, lane);
  __syncthreads();
  for (int e = tid; e < 128 * 128; e += 256) {
    const int l = e >> 7, c = e & 127;
    vt[l * LDAH + c] = (_Float16)((float)vt[l * LDAH + c] * pk[c]);
  }
  __syncthreads();

  // ---- kvt = kv@wt+bt + q (residual) -> nx buffer; att = kvt@wo+bo ----
  gemm128(vt, w16 + WT_T, bt, nx, qt, wave, lane);
  __syncthreads();
  gemm128(nx, w16 + WO_T, bo, att, nullptr, wave, lane);
  __syncthreads();

  // ---- window aggregation + agg_b + x residual -> fp32 att_out ----
  agg_fn(att, w16 + AGG_T, aggb, x, aout, n, t, wave, lane);
  __syncthreads();

  // ---- LN2 over channels per (v); write f16 [32][136] (pad rows zero) ----
  for (int rep = 0; rep < 4; ++rep) {
    const int v = wave + rep * 8;
    const float* rp = aout + v * LDXS;
    float s = 0.f, s2 = 0.f;
    if (v < 25) {
#pragma unroll
      for (int j = 0; j < 4; ++j) { const float a = rp[lane * 4 + j]; s += a; s2 += a * a; }
    }
#pragma unroll
    for (int o = 16; o; o >>= 1) { s += __shfl_xor(s, o, 32); s2 += __shfl_xor(s2, o, 32); }
    const float mean = s * (1.f / 128.f);
    const float var = s2 * (1.f / 128.f) - mean * mean;
    const float inv = rsqrtf(var + 1e-5f);
    _Float16* op = ln2h + v * LDAH;
#pragma unroll
    for (int j = 0; j < 4; ++j) {
      const int c = lane * 4 + j;
      const float val = (v < 25) ? (rp[c] - mean) * inv * g2[c] + b2[c] : 0.f;
      op[c] = (_Float16)val;
    }
  }
  __syncthreads();

  // ---- FFN: h1 = gelu(ln2@w1+b1) ; out = h1@w2+b2 + att_out ----
  ffn1_fn(ln2h, w16 + F1_T, fb1, h1, wave, lane);
  __syncthreads();
  ffn2_fn(h1, w16 + F2_T, fb2, aout, out, n, t, wave, lane);
}

extern "C" void kernel_launch(void* const* d_in, const int* in_sizes, int n_in,
                              void* d_out, int out_size, void* d_ws, size_t ws_size,
                              hipStream_t stream) {
  (void)in_sizes; (void)n_in; (void)out_size; (void)ws_size;
  const float* x    = (const float*)d_in[0];
  const float* g1   = (const float*)d_in[1];
  const float* b1   = (const float*)d_in[2];
  const float* wq   = (const float*)d_in[3];
  const float* bq   = (const float*)d_in[4];
  const float* wqa  = (const float*)d_in[5];
  const float* bqa  = (const float*)d_in[6];
  const float* wk   = (const float*)d_in[7];
  const float* bk   = (const float*)d_in[8];
  const float* wka  = (const float*)d_in[9];
  const float* bka  = (const float*)d_in[10];
  const float* wv   = (const float*)d_in[11];
  const float* bv   = (const float*)d_in[12];
  const float* wt   = (const float*)d_in[13];
  const float* bt   = (const float*)d_in[14];
  const float* wo   = (const float*)d_in[15];
  const float* bo   = (const float*)d_in[16];
  const float* aggw = (const float*)d_in[17];
  const float* aggb = (const float*)d_in[18];
  const float* g2   = (const float*)d_in[19];
  const float* b2   = (const float*)d_in[20];
  const float* f1   = (const float*)d_in[21];
  const float* fb1  = (const float*)d_in[22];
  const float* f2   = (const float*)d_in[23];
  const float* fb2  = (const float*)d_in[24];

  _Float16* w16 = (_Float16*)d_ws;

  convert_weights<<<(W16_TOTAL + 255) / 256, 256, 0, stream>>>(
      wq, wk, wv, wt, wo, f1, f2, wqa, wka, aggw, w16);

  fused_block_kernel<<<32 * 120, 256, SMEM_TOTAL, stream>>>(
      x, g1, b1, bq, bqa, bk, bka, bv, bt, bo, aggb, g2, b2, fb1, fb2, w16,
      (float*)d_out);
}
